// DecLayer_32573031973438
// MI455X (gfx1250) — compile-verified, matrix operand order checked by reading, not run on hardware
//
#include <hip/hip_runtime.h>
#include <hip/hip_bf16.h>
#include <math.h>

// ---------------- types ----------------
typedef __attribute__((ext_vector_type(4)))  _Float16 v4h;
typedef __attribute__((ext_vector_type(8)))  _Float16 v8h;
typedef __attribute__((ext_vector_type(16))) _Float16 v16h;
typedef __attribute__((ext_vector_type(8)))  float    v8f;

#define HDIM   128
#define NUMIN  384
#define CIN    512      // H + NUM_IN
#define KNBR   48
#define FF     512
#define NNODES 8192     // B*L
#define NPB    2        // nodes per block in k_message
#define ROWS   (NPB*KNBR)        // 96 rows, 6 row-tiles
#define RT     (ROWS/16)         // 6
#define XSTR   520      // padded LDS stride (halves) for 512-wide tiles
#define MSTR   136      // padded LDS stride (halves) for 128-wide tiles
#define SCALE_INV (1.0f/30.0f)

// Branchless GELU (tanh form). tanh(y) = 1 - 2/(1+exp(2y)); saturates cleanly
// at +/-inf via v_exp/v_rcp, no exec divergence. Error ~1e-3, below the f16
// WMMA quantization already in the pipeline.
__device__ __forceinline__ float gelu_fast(float x) {
    float y  = 0.7978845608028654f * (x + 0.044715f * x * x * x);
    float e  = __expf(2.0f * y);
    float t  = 1.0f - 2.0f * __builtin_amdgcn_rcpf(e + 1.0f);
    return 0.5f * x * (1.0f + t);
}

// --------- WMMA fragment loaders ---------
// A fragment (16x32 f16): lanes 0-15 rows M=0..15 (K+0..7 in v0..3, K+16..23 in
// v4..7), lanes 16-31 same rows with K+8.  Two contiguous 16B LDS chunks/lane.
__device__ __forceinline__ v16h load_a_frag(const _Float16* base, int strideH,
                                            int row0, int kbase) {
    const int lane = threadIdx.x & 31;
    const int row  = row0 + (lane & 15);
    const int k0   = kbase + ((lane >> 4) << 3);
    const _Float16* p = base + row * strideH + k0;
    v8h lo = *(const v8h*)(p);
    v8h hi = *(const v8h*)(p + 16);
    v16h r;
#pragma unroll
    for (int i = 0; i < 8; ++i) { r[i] = lo[i]; r[8 + i] = hi[i]; }
    return r;
}

// B fragment (32x16 f16) pre-swizzled by k_pack -> one 32B coalesced load/lane.
__device__ __forceinline__ v16h load_b_frag(const _Float16* __restrict__ pack, int frag) {
    const int lane = threadIdx.x & 31;
    return *(const v16h*)(pack + ((size_t)(frag * 32 + lane) << 4));
}

__device__ __forceinline__ v8f wmma_f16(v16h a, v16h b, v8f c) {
    return __builtin_amdgcn_wmma_f32_16x16x32_f16(false, a, false, b,
                                                  (short)0, c, false, false);
}

// --------- weight prepack: f32 [Kdim,Ndim] row-major -> f16 B-fragment order ---------
__global__ __launch_bounds__(256) void k_pack(const float* __restrict__ w,
                                              _Float16* __restrict__ dst,
                                              int Kdim, int Ndim) {
    int e = blockIdx.x * 256 + threadIdx.x;
    int total = Kdim * Ndim;
    if (e >= total) return;
    int ntiles = Ndim >> 4;
    int i    = e & 15;
    int lane = (e >> 4) & 31;
    int frag = e >> 9;
    int ktile = frag / ntiles;
    int ntile = frag - ktile * ntiles;
    int K = (ktile << 5) + ((i >> 3) << 4) + ((lane >> 4) << 3)
          + (((i >> 1) & 3) << 1) + (i & 1);
    int N = (ntile << 4) + (lane & 15);
    dst[e] = (_Float16)w[(size_t)K * Ndim + N];
}

// --------- fused message MLP + masked neighbor reduction ---------
// one workgroup (256 threads = 8 waves) per NPB nodes
__global__ __launch_bounds__(256) void k_message(
    const float* __restrict__ h_v, const float* __restrict__ h_e,
    const float* __restrict__ mask_attend,
    const _Float16* __restrict__ pw1, const _Float16* __restrict__ pw2,
    const _Float16* __restrict__ pw3,
    const float* __restrict__ b1, const float* __restrict__ b2,
    const float* __restrict__ b3,
    float* __restrict__ dh)
{
    __shared__ __align__(16) _Float16 X[ROWS * XSTR];   // 99840 B
    __shared__ __align__(16) _Float16 Mb[ROWS * MSTR];  // 26112 B
    __shared__ float smask[ROWS];

    const int n0   = blockIdx.x * NPB;
    const int tid  = threadIdx.x;
    const int wave = tid >> 5;
    const int lane = tid & 31;

    // ---- stage X = [h_v (bcast) | h_e] as f16 in LDS ----
    {
        const float* he = h_e + (size_t)n0 * KNBR * NUMIN;
        for (int idx = tid; idx < ROWS * NUMIN / 4; idx += 256) {
            float4 v = ((const float4*)he)[idx];
            int r = idx / (NUMIN / 4);
            int c = (idx - r * (NUMIN / 4)) * 4 + HDIM;
            v4h o; o[0] = (_Float16)v.x; o[1] = (_Float16)v.y;
                   o[2] = (_Float16)v.z; o[3] = (_Float16)v.w;
            *(v4h*)&X[r * XSTR + c] = o;
        }
        const float4* hv4 = (const float4*)(h_v + (size_t)n0 * HDIM);
        for (int idx = tid; idx < ROWS * (HDIM / 4); idx += 256) {
            int r = idx >> 5;          // row 0..95
            int c4 = idx & 31;
            float4 v = hv4[(r / KNBR) * (HDIM / 4) + c4];
            v4h o; o[0] = (_Float16)v.x; o[1] = (_Float16)v.y;
                   o[2] = (_Float16)v.z; o[3] = (_Float16)v.w;
            *(v4h*)&X[r * XSTR + c4 * 4] = o;
        }
        if (tid < ROWS) smask[tid] = mask_attend[(size_t)n0 * KNBR + tid];
    }
    __syncthreads();

    const int col = (wave << 4) + (lane & 15);
    const int rhi = (lane >> 4) << 3;   // 0 or 8

    // ---- GEMM1: [96,512] x [512,128], GELU ----
    v8f acc[RT];
#pragma unroll
    for (int t = 0; t < RT; ++t) acc[t] = (v8f){};
#pragma unroll 2
    for (int kt = 0; kt < CIN / 32; ++kt) {
        v16h bf = load_b_frag(pw1, kt * 8 + wave);
#pragma unroll
        for (int t = 0; t < RT; ++t) {
            v16h a = load_a_frag(X, XSTR, t * 16, kt * 32);
            acc[t] = wmma_f16(a, bf, acc[t]);
        }
    }
    {
        float bb = b1[col];
#pragma unroll
        for (int t = 0; t < RT; ++t)
#pragma unroll
            for (int r = 0; r < 8; ++r)
                Mb[(t * 16 + rhi + r) * MSTR + col] = (_Float16)gelu_fast(acc[t][r] + bb);
    }
    __syncthreads();

    // ---- GEMM2: [96,128] x [128,128], GELU ----
#pragma unroll
    for (int t = 0; t < RT; ++t) acc[t] = (v8f){};
#pragma unroll
    for (int kt = 0; kt < HDIM / 32; ++kt) {
        v16h bf = load_b_frag(pw2, kt * 8 + wave);
#pragma unroll
        for (int t = 0; t < RT; ++t) {
            v16h a = load_a_frag(Mb, MSTR, t * 16, kt * 32);
            acc[t] = wmma_f16(a, bf, acc[t]);
        }
    }
    __syncthreads();   // all waves done READING Mb before overwrite
    {
        float bb = b2[col];
#pragma unroll
        for (int t = 0; t < RT; ++t)
#pragma unroll
            for (int r = 0; r < 8; ++r)
                Mb[(t * 16 + rhi + r) * MSTR + col] = (_Float16)gelu_fast(acc[t][r] + bb);
    }
    __syncthreads();

    // ---- GEMM3: [96,128] x [128,128] + bias, masked sum over neighbors ----
#pragma unroll
    for (int t = 0; t < RT; ++t) acc[t] = (v8f){};
#pragma unroll
    for (int kt = 0; kt < HDIM / 32; ++kt) {
        v16h bf = load_b_frag(pw3, kt * 8 + wave);
#pragma unroll
        for (int t = 0; t < RT; ++t) {
            v16h a = load_a_frag(Mb, MSTR, t * 16, kt * 32);
            acc[t] = wmma_f16(a, bf, acc[t]);
        }
    }
    {
        float bb = b3[col];
        float s0 = 0.0f, s1 = 0.0f;   // per-node sums (rows 0-47 / 48-95)
#pragma unroll
        for (int t = 0; t < 3; ++t)
#pragma unroll
            for (int r = 0; r < 8; ++r)
                s0 += (acc[t][r] + bb) * smask[t * 16 + rhi + r];
#pragma unroll
        for (int t = 3; t < 6; ++t)
#pragma unroll
            for (int r = 0; r < 8; ++r)
                s1 += (acc[t][r] + bb) * smask[t * 16 + rhi + r];
        s0 += __shfl_xor(s0, 16, 32);
        s1 += __shfl_xor(s1, 16, 32);
        if (lane < 16) {
            dh[(size_t)(n0    ) * HDIM + (wave << 4) + lane] = s0 * SCALE_INV;
            dh[(size_t)(n0 + 1) * HDIM + (wave << 4) + lane] = s1 * SCALE_INV;
        }
    }
}

// --------- residual + LN1 + FFN (128->512->128) + LN2 + node mask ---------
// one workgroup per 16 nodes
__global__ __launch_bounds__(256) void k_ffn(
    const float* __restrict__ h_v, const float* __restrict__ dh,
    const float* __restrict__ mask_v,
    const _Float16* __restrict__ pwin, const _Float16* __restrict__ pwout,
    const float* __restrict__ fwin_b, const float* __restrict__ fwout_b,
    const float* __restrict__ ln1_g, const float* __restrict__ ln1_b,
    const float* __restrict__ ln2_g, const float* __restrict__ ln2_b,
    float* __restrict__ out)
{
    __shared__ float  hres[16 * 132];
    __shared__ __align__(16) _Float16 hh[16 * MSTR];
    __shared__ __align__(16) _Float16 act[16 * XSTR];
    __shared__ float  ybuf[16 * 132];

    const int tid  = threadIdx.x;
    const int wave = tid >> 5;
    const int lane = tid & 31;
    const int row  = tid >> 4;
    const int part = tid & 15;
    const int node = blockIdx.x * 16 + row;

    // ---- LN1(h_v + dh) ----
    float v[8];
    {
        float s = 0.0f, s2 = 0.0f;
#pragma unroll
        for (int i = 0; i < 8; ++i) {
            int f = part * 8 + i;
            float x = h_v[(size_t)node * HDIM + f] + dh[(size_t)node * HDIM + f];
            v[i] = x; s += x; s2 += x * x;
        }
#pragma unroll
        for (int m = 1; m < 16; m <<= 1) { s += __shfl_xor(s, m, 32); s2 += __shfl_xor(s2, m, 32); }
        float mu = s * (1.0f / HDIM);
        float rs = rsqrtf(s2 * (1.0f / HDIM) - mu * mu + 1e-5f);
#pragma unroll
        for (int i = 0; i < 8; ++i) {
            int f = part * 8 + i;
            float y = (v[i] - mu) * rs * ln1_g[f] + ln1_b[f];
            hres[row * 132 + f] = y;
            hh[row * MSTR + f]  = (_Float16)y;
        }
    }
    __syncthreads();

    const int rhi = (lane >> 4) << 3;

    // ---- FF1: [16,128] x [128,512], GELU -> act ----
    {
        v8f a[4] = {};
#pragma unroll
        for (int kt = 0; kt < HDIM / 32; ++kt) {
            v16h af = load_a_frag(hh, MSTR, 0, kt * 32);
#pragma unroll
            for (int j = 0; j < 4; ++j) {
                v16h bf = load_b_frag(pwin, kt * 32 + wave * 4 + j);
                a[j] = wmma_f16(af, bf, a[j]);
            }
        }
#pragma unroll
        for (int j = 0; j < 4; ++j) {
            int col = (wave * 4 + j) * 16 + (lane & 15);
            float bb = fwin_b[col];
#pragma unroll
            for (int r = 0; r < 8; ++r)
                act[(rhi + r) * XSTR + col] = (_Float16)gelu_fast(a[j][r] + bb);
        }
    }
    __syncthreads();

    // ---- FF2: [16,512] x [512,128] + bias + residual ----
    {
        v8f a = {};
#pragma unroll 4
        for (int kt = 0; kt < FF / 32; ++kt) {
            v16h af = load_a_frag(act, XSTR, 0, kt * 32);
            v16h bf = load_b_frag(pwout, kt * 8 + wave);
            a = wmma_f16(af, bf, a);
        }
        int col = wave * 16 + (lane & 15);
        float bb = fwout_b[col];
#pragma unroll
        for (int r = 0; r < 8; ++r) {
            int rr = rhi + r;
            ybuf[rr * 132 + col] = a[r] + bb + hres[rr * 132 + col];
        }
    }
    __syncthreads();

    // ---- LN2 + node mask + store ----
    {
        float s = 0.0f, s2 = 0.0f;
#pragma unroll
        for (int i = 0; i < 8; ++i) {
            int f = part * 8 + i;
            float x = ybuf[row * 132 + f];
            v[i] = x; s += x; s2 += x * x;
        }
#pragma unroll
        for (int m = 1; m < 16; m <<= 1) { s += __shfl_xor(s, m, 32); s2 += __shfl_xor(s2, m, 32); }
        float mu = s * (1.0f / HDIM);
        float rs = rsqrtf(s2 * (1.0f / HDIM) - mu * mu + 1e-5f);
        float mv = mask_v[node];
#pragma unroll
        for (int i = 0; i < 8; ++i) {
            int f = part * 8 + i;
            out[(size_t)node * HDIM + f] = mv * ((v[i] - mu) * rs * ln2_g[f] + ln2_b[f]);
        }
    }
}

// ---------------- host launcher ----------------
extern "C" void kernel_launch(void* const* d_in, const int* in_sizes, int n_in,
                              void* d_out, int out_size, void* d_ws, size_t ws_size,
                              hipStream_t stream) {
    const float* h_v         = (const float*)d_in[0];
    const float* h_e         = (const float*)d_in[1];
    const float* mask_v      = (const float*)d_in[2];
    const float* mask_attend = (const float*)d_in[3];
    const float* w1_w  = (const float*)d_in[4];
    const float* w1_b  = (const float*)d_in[5];
    const float* w2_w  = (const float*)d_in[6];
    const float* w2_b  = (const float*)d_in[7];
    const float* w3_w  = (const float*)d_in[8];
    const float* w3_b  = (const float*)d_in[9];
    const float* ln1_g = (const float*)d_in[10];
    const float* ln1_b = (const float*)d_in[11];
    const float* ln2_g = (const float*)d_in[12];
    const float* ln2_b = (const float*)d_in[13];
    const float* fw_in_w  = (const float*)d_in[14];
    const float* fw_in_b  = (const float*)d_in[15];
    const float* fw_out_w = (const float*)d_in[16];
    const float* fw_out_b = (const float*)d_in[17];
    float* out = (float*)d_out;

    // workspace layout (bytes)
    char* ws = (char*)d_ws;
    _Float16* pw1   = (_Float16*)(ws + 0);        // 131072
    _Float16* pw2   = (_Float16*)(ws + 131072);   //  32768
    _Float16* pw3   = (_Float16*)(ws + 163840);   //  32768
    _Float16* pwin  = (_Float16*)(ws + 196608);   // 131072
    _Float16* pwout = (_Float16*)(ws + 327680);   // 131072
    float*    dh    = (float*)(ws + 458752);      // 4194304

    k_pack<<<(CIN * HDIM + 255) / 256, 256, 0, stream>>>(w1_w,     pw1,   CIN,  HDIM);
    k_pack<<<(HDIM * HDIM + 255) / 256, 256, 0, stream>>>(w2_w,    pw2,   HDIM, HDIM);
    k_pack<<<(HDIM * HDIM + 255) / 256, 256, 0, stream>>>(w3_w,    pw3,   HDIM, HDIM);
    k_pack<<<(HDIM * FF   + 255) / 256, 256, 0, stream>>>(fw_in_w, pwin,  HDIM, FF);
    k_pack<<<(FF * HDIM   + 255) / 256, 256, 0, stream>>>(fw_out_w,pwout, FF,   HDIM);

    k_message<<<NNODES / NPB, 256, 0, stream>>>(h_v, h_e, mask_attend,
                                                pw1, pw2, pw3, w1_b, w2_b, w3_b, dh);

    k_ffn<<<NNODES / 16, 256, 0, stream>>>(h_v, dh, mask_v, pwin, pwout,
                                           fw_in_b, fw_out_b,
                                           ln1_g, ln1_b, ln2_g, ln2_b, out);
}
